// DNALayer_48601849921697
// MI455X (gfx1250) — matrix operand order repaired and checked by reading, not compiled, not run on hardware
//
#include <hip/hip_runtime.h>
#include <hip/hip_bf16.h>
#include <math.h>
#include <stdint.h>

// ---------------- problem constants (match reference) ----------------
#define HID   1024
#define NEXP  8
#define DFF   4096
#define NTOK  4096      // B*S = 2*2048
#define CAP   4096      // max tokens per expert (top-2 experts are distinct)

// LDS row strides (padded, halves) — keep 16B alignment, avoid bank conflicts
#define XAP   1032      // xA row stride   (1024 + 8) ; 2064 B = 129*16
#define HMP   136       // hmid row stride (128 + 8)  ; 272 B  = 17*16
#define WTP   40        // wT row stride   (32 + 8)   ; 80 B   = 5*16

typedef __attribute__((ext_vector_type(16))) _Float16 v16h;
typedef __attribute__((ext_vector_type(8)))  float    v8f;
typedef __attribute__((ext_vector_type(4)))  int      v4i;

#define AS_GLOBAL __attribute__((address_space(1)))
#define AS_LOCAL  __attribute__((address_space(3)))

typedef AS_GLOBAL v4i* gv4i_p;   // int4 in global address space
typedef AS_LOCAL  v4i* lv4i_p;   // int4 in LDS address space

union AF { uint4 u[2]; v16h v; };

// ----------------- CDNA5 async global->LDS copy path -----------------
#if defined(__has_builtin)
#if __has_builtin(__builtin_amdgcn_global_load_async_to_lds_b128) && \
    __has_builtin(__builtin_amdgcn_s_wait_asynccnt)
#define USE_ASYNC_BUILTIN 1
#endif
#endif

__device__ __forceinline__ void async_copy_b128(const float* gp, float* lp) {
#if defined(USE_ASYNC_BUILTIN)
    __builtin_amdgcn_global_load_async_to_lds_b128(
        (gv4i_p)gp, (lv4i_p)lp, 0, 0);
#else
    unsigned int loff = (unsigned int)(uintptr_t)(AS_LOCAL void*)lp;
    asm volatile("global_load_async_to_lds_b128 %0, %1, off"
                 :: "v"(loff), "v"((unsigned long long)(uintptr_t)gp)
                 : "memory");
#endif
}

template <int N>
__device__ __forceinline__ void wait_async() {
#if defined(USE_ASYNC_BUILTIN)
    __builtin_amdgcn_s_wait_asynccnt(N);
#else
    asm volatile("s_wait_asynccnt %0" :: "i"(N) : "memory");
#endif
}

// branchless tanh-GELU: one v_exp_f32 + a few VALU, no EXEC divergence
__device__ __forceinline__ float gelu_tanh(float v) {
    const float c = 0.7978845608028654f; // sqrt(2/pi)
    float z = c * (v + 0.044715f * v * v * v);
    z = fminf(fmaxf(z, -9.0f), 9.0f);
    float u = __expf(2.0f * z);
    float t = (u - 1.0f) / (u + 1.0f);
    return 0.5f * v * (1.0f + t);
}

// ---------------------------------------------------------------------
// Kernel 0: zero output buffer (we atomically combine into it) + counters
// ---------------------------------------------------------------------
__global__ void zero_kernel(float* __restrict__ out, size_t n, int* __restrict__ counts) {
    size_t i = (size_t)blockIdx.x * blockDim.x + threadIdx.x;
    size_t stride = (size_t)gridDim.x * blockDim.x;
    for (; i < n; i += stride) out[i] = 0.0f;
    if (blockIdx.x == 0 && threadIdx.x < NEXP) counts[threadIdx.x] = 0;
}

// ---------------------------------------------------------------------
// Kernel 1: router — one wave32 per token
// ---------------------------------------------------------------------
__global__ __launch_bounds__(256) void router_kernel(
    const float* __restrict__ x, const float* __restrict__ Wr,
    const float* __restrict__ br, int* __restrict__ counts,
    int* __restrict__ etok, float* __restrict__ ewt)
{
    const int lane = threadIdx.x & 31;
    const int wave = threadIdx.x >> 5;
    const int tok  = blockIdx.x * 8 + wave;   // 8 waves per 256-thread block

    float acc[NEXP];
    #pragma unroll
    for (int e = 0; e < NEXP; ++e) acc[e] = 0.0f;

    const float* xr = x + (size_t)tok * HID;
    for (int h = lane; h < HID; h += 32) {     // coalesced across the wave
        float xv = xr[h];
        #pragma unroll
        for (int e = 0; e < NEXP; ++e) acc[e] += xv * Wr[h * NEXP + e];
    }
    #pragma unroll
    for (int e = 0; e < NEXP; ++e) {
        #pragma unroll
        for (int off = 16; off > 0; off >>= 1)
            acc[e] += __shfl_xor(acc[e], off, 32);
    }

    if (lane == 0) {
        float l[NEXP];
        float mx = -3.4e38f;
        #pragma unroll
        for (int e = 0; e < NEXP; ++e) { l[e] = acc[e] + br[e]; mx = fmaxf(mx, l[e]); }
        float p[NEXP]; float sum = 0.0f;
        #pragma unroll
        for (int e = 0; e < NEXP; ++e) { p[e] = __expf(l[e] - mx); sum += p[e]; }
        int i0 = 0;
        #pragma unroll
        for (int e = 1; e < NEXP; ++e) if (p[e] > p[i0]) i0 = e;
        int i1 = (i0 == 0) ? 1 : 0;
        #pragma unroll
        for (int e = 0; e < NEXP; ++e) if (e != i0 && p[e] > p[i1]) i1 = e;
        float w0 = p[i0] / sum, w1 = p[i1] / sum;
        float s = w0 + w1 + 1e-8f;
        w0 /= s; w1 /= s;
        int p0 = atomicAdd(&counts[i0], 1);
        etok[i0 * CAP + p0] = tok;  ewt[i0 * CAP + p0] = w0;
        int p1 = atomicAdd(&counts[i1], 1);
        etok[i1 * CAP + p1] = tok;  ewt[i1 * CAP + p1] = w1;
    }
}

// ---------------------------------------------------------------------
// Kernel 2: fused gathered expert MLP, f16 WMMA, async-DMA weight tiles
// ---------------------------------------------------------------------
// issue async copy of one 32x128 fp32 weight tile (row stride ld) -> raw LDS
__device__ __forceinline__ void issue_tile(const float* Wp, size_t ld,
                                           float* raw, int tid) {
    #pragma unroll
    for (int q = 0; q < 4; ++q) {
        int c = tid + q * 256;       // 16-byte chunk id (1024 chunks / tile)
        int k = c >> 5;              // row (32 chunks per 128-float row)
        int n = (c & 31) * 4;        // column
        async_copy_b128(Wp + (size_t)k * ld + n, raw + c * 4);
    }
}

// convert raw fp32 tile (row-major [32][128]) -> transposed padded f16 wT[n][k].
// Each thread owns a 4x4 (k x n) block: 4 x ds_load_b128 + 4 x packed b64 store.
__device__ __forceinline__ void convert_tile(const float* raw, _Float16* wTp, int tid) {
    const int kb = (tid >> 5) * 4;       // k block base (wave-uniform)
    const int nb = (tid & 31) * 4;       // n block base
    float4 r0 = *(const float4*)&raw[(kb + 0) * 128 + nb];
    float4 r1 = *(const float4*)&raw[(kb + 1) * 128 + nb];
    float4 r2 = *(const float4*)&raw[(kb + 2) * 128 + nb];
    float4 r3 = *(const float4*)&raw[(kb + 3) * 128 + nb];
    union { _Float16 h[4]; uint2 u; } c0, c1, c2, c3;
    c0.h[0] = (_Float16)r0.x; c0.h[1] = (_Float16)r1.x;
    c0.h[2] = (_Float16)r2.x; c0.h[3] = (_Float16)r3.x;
    c1.h[0] = (_Float16)r0.y; c1.h[1] = (_Float16)r1.y;
    c1.h[2] = (_Float16)r2.y; c1.h[3] = (_Float16)r3.y;
    c2.h[0] = (_Float16)r0.z; c2.h[1] = (_Float16)r1.z;
    c2.h[2] = (_Float16)r2.z; c2.h[3] = (_Float16)r3.z;
    c3.h[0] = (_Float16)r0.w; c3.h[1] = (_Float16)r1.w;
    c3.h[2] = (_Float16)r2.w; c3.h[3] = (_Float16)r3.w;
    *(uint2*)&wTp[(nb + 0) * WTP + kb] = c0.u;   // 8B-aligned: nb*80 + kb*2
    *(uint2*)&wTp[(nb + 1) * WTP + kb] = c1.u;
    *(uint2*)&wTp[(nb + 2) * WTP + kb] = c2.u;
    *(uint2*)&wTp[(nb + 3) * WTP + kb] = c3.u;
}

__global__ __launch_bounds__(256, 1) void moe_gemm_kernel(
    const float* __restrict__ x,  const float* __restrict__ W1,
    const float* __restrict__ b1, const float* __restrict__ W2,
    const float* __restrict__ b2, const int* __restrict__ counts,
    const int* __restrict__ etok, const float* __restrict__ ewt,
    float* __restrict__ out)
{
    const int e  = blockIdx.y;
    const int mt = blockIdx.x;
    const int cnt = counts[e];
    if (mt * 16 >= cnt) return;            // uniform early-exit

    __shared__ __align__(16) _Float16 xA[16 * XAP];     // gathered x tile (f16)
    __shared__ __align__(16) _Float16 hmid[16 * HMP];   // gelu(x@W1) chunk (f16)
    __shared__ __align__(16) _Float16 wT[128 * WTP];    // transposed weight tile
    __shared__ __align__(16) float    raw[2][32 * 128]; // async landing buffers
    __shared__ int   stok[16];
    __shared__ float swt[16];

    const int tid  = threadIdx.x;
    const int lane = tid & 31;
    const int wave = tid >> 5;
    const int ln16 = lane & 15;
    const int kg   = lane >> 4;            // lane half-group (0/1)

    if (tid < 16) {
        int gi = mt * 16 + tid;
        int tk = 0; float w = 0.0f;
        if (gi < cnt) { tk = etok[e * CAP + gi]; w = ewt[e * CAP + gi]; }
        stok[tid] = tk; swt[tid] = w;      // padded rows: token 0, weight 0
    }
    __syncthreads();

    // gather 16 token rows of x -> LDS f16 (vectorized, coalesced)
    for (int i4 = tid * 4; i4 < 16 * HID; i4 += 1024) {
        int r = i4 >> 10;
        int h = i4 & (HID - 1);
        float4 v = *(const float4*)&x[(size_t)stok[r] * HID + h];
        _Float16* d = &xA[r * XAP + h];
        d[0] = (_Float16)v.x; d[1] = (_Float16)v.y;
        d[2] = (_Float16)v.z; d[3] = (_Float16)v.w;
    }

    const v8f zero = {};
    v8f accy[8];
    #pragma unroll
    for (int i = 0; i < 8; ++i) accy[i] = zero;

    const size_t w1base = (size_t)e * HID * DFF;
    const size_t w2base = (size_t)e * DFF * HID;

    #pragma unroll 1
    for (int dc = 0; dc < DFF / 128; ++dc) {
        const int fbase = dc * 128;

        // ---------- stage 1: hmid[16 x 128] = gelu(xA @ W1 + b1) ----------
        v8f acch = zero;
        issue_tile(W1 + w1base + fbase, DFF, raw[0], tid);   // prologue tile 0
        #pragma unroll 1
        for (int ks = 0; ks < HID / 32; ++ks) {
            const int cur = ks & 1, nxt = cur ^ 1;
            if (ks + 1 < HID / 32) {
                issue_tile(W1 + w1base + (size_t)((ks + 1) * 32) * DFF + fbase,
                           DFF, raw[nxt], tid);
                wait_async<4>();           // tile `cur` landed (in-order)
            } else {
                wait_async<0>();
            }
            __syncthreads();               // all waves' chunks of raw[cur] done
            convert_tile(raw[cur], wT, tid);
            __syncthreads();               // wT ready
            AF a, b;
            const _Float16* ap = &xA[ln16 * XAP + ks * 32 + kg * 8];
            a.u[0] = *(const uint4*)ap;          // k = base + kg*8 + 0..7
            a.u[1] = *(const uint4*)(ap + 16);   // k = base + 16 + kg*8 + 0..7
            const _Float16* bp = &wT[(wave * 16 + ln16) * WTP + kg * 8];
            b.u[0] = *(const uint4*)bp;          // k = kg*8 + 0..7
            b.u[1] = *(const uint4*)(bp + 16);   // k = 16 + kg*8 + 0..7
            acch = __builtin_amdgcn_wmma_f32_16x16x32_f16(
                       false, a.v, false, b.v, (short)0, acch, false, false);
        }
        __syncthreads();
        {   // GELU + b1 epilogue -> hmid (f16)
            int col = wave * 16 + ln16;
            float b1v = b1[e * DFF + fbase + col];
            #pragma unroll
            for (int r = 0; r < 8; ++r) {
                int m = r + kg * 8;              // C layout: VGPR r, lane-half kg
                hmid[m * HMP + col] = (_Float16)gelu_tanh(acch[r] + b1v);
            }
        }
        __syncthreads();

        // ---------- stage 2: accy += hmid @ W2[fbase:fbase+128, :] ----------
        // 32 tiles: t -> (nt = t>>2 slice of H, ks2 = t&3 k-step)
        issue_tile(W2 + w2base + (size_t)fbase * HID, HID, raw[0], tid);
        #pragma unroll 1
        for (int t = 0; t < 32; ++t) {
            const int cur = t & 1, nxt = cur ^ 1;
            if (t + 1 < 32) {
                int nt2 = (t + 1) >> 2, ks2n = (t + 1) & 3;
                issue_tile(W2 + w2base + (size_t)(fbase + ks2n * 32) * HID + nt2 * 128,
                           HID, raw[nxt], tid);
                wait_async<4>();
            } else {
                wait_async<0>();
            }
            __syncthreads();
            convert_tile(raw[cur], wT, tid);
            __syncthreads();
            const int nt = t >> 2, ks2 = t & 3;
            AF a, b;
            const _Float16* ap = &hmid[ln16 * HMP + ks2 * 32 + kg * 8];
            a.u[0] = *(const uint4*)ap;
            a.u[1] = *(const uint4*)(ap + 16);
            const _Float16* bp = &wT[(wave * 16 + ln16) * WTP + kg * 8];
            b.u[0] = *(const uint4*)bp;
            b.u[1] = *(const uint4*)(bp + 16);
            accy[nt] = __builtin_amdgcn_wmma_f32_16x16x32_f16(
                           false, a.v, false, b.v, (short)0, accy[nt], false, false);
        }
    }

    // ---------- epilogue: +b2, x routing weight, atomic combine ----------
    #pragma unroll 1
    for (int nt = 0; nt < 8; ++nt) {
        int col = nt * 128 + wave * 16 + ln16;
        float b2v = b2[e * HID + col];
        #pragma unroll
        for (int r = 0; r < 8; ++r) {
            int m = r + kg * 8;
            float val = (accy[nt][r] + b2v) * swt[m];
            atomicAdd(&out[(size_t)stok[m] * HID + col], val);
        }
    }
}

// ---------------------------------------------------------------------
extern "C" void kernel_launch(void* const* d_in, const int* in_sizes, int n_in,
                              void* d_out, int out_size, void* d_ws, size_t ws_size,
                              hipStream_t stream) {
    const float* x  = (const float*)d_in[0];
    const float* Wr = (const float*)d_in[1];
    const float* br = (const float*)d_in[2];
    const float* W1 = (const float*)d_in[3];
    const float* b1 = (const float*)d_in[4];
    const float* W2 = (const float*)d_in[5];
    const float* b2 = (const float*)d_in[6];
    (void)in_sizes; (void)n_in; (void)ws_size;
    float* out = (float*)d_out;

    char* ws = (char*)d_ws;
    int*   counts = (int*)ws;                                      // 32 B (pad 256)
    int*   etok   = (int*)(ws + 256);                              // 128 KB
    float* ewt    = (float*)(ws + 256 + NEXP * CAP * sizeof(int)); // 128 KB

    zero_kernel<<<4096, 256, 0, stream>>>(out, (size_t)out_size, counts);
    router_kernel<<<NTOK / 8, 256, 0, stream>>>(x, Wr, br, counts, etok, ewt);
    dim3 grid(NTOK / 16, NEXP);
    moe_gemm_kernel<<<grid, 256, 0, stream>>>(x, W1, b1, W2, b2,
                                              counts, etok, ewt, out);
}
